// DecentralPlannerNet_13262859010079
// MI455X (gfx1250) — matrix operand order repaired and checked by reading, not compiled, not run on hardware
//
#include <hip/hip_runtime.h>
#include <hip/hip_bf16.h>

typedef __attribute__((ext_vector_type(16))) _Float16 v16h;
typedef __attribute__((ext_vector_type(8)))  _Float16 v8h;
typedef __attribute__((ext_vector_type(8)))  float    v8f;

#define NWAVES 4      // waves per CNN block
#define IMG    4      // images per CNN block
#define BN_TOT 10240  // B*N images
#define FEAT   128

// ---- WMMA fragment index helpers (this file's contiguous k order, not the ISA table order) ----
// A (16x32, MxK): lane -> m = lane&15 ; element j -> K = ((lane>>4)<<4) + j (contiguous)
static __device__ __forceinline__ int ak_idx(int j, int lane) {
    return ((lane >> 4) << 4) | j;
}
// B (32x16, KxN): lane -> n = lane&15 ; lanes 16-31 hold K=16..31.
// K = kc*32 + ((lane>>4)<<4) + j  ==> contiguous in j (vectorizable loads).
// C/D (16x16 f32): element r, lane -> row = r + 8*(lane>=16), col = lane&15

// ------------------------------------------------------------------
// Workspace layout (bytes) -- all 32B aligned
// Conv im2col K-ordering: layers 1-4: k = (rr*3+cc)*CIN + cin (HWC)
//                         layer 0 (ch-padded to 8): k = rr*32 + cc*8 + cin
// ------------------------------------------------------------------
#define WS_W16    0            // 141312 halves: conv weights, FRAGMENT order
#define WS_SCALE  282624       // 320 f32 scale
#define WS_SHIFT  283904       // 320 f32 shift
#define WS_MW16   285184       // 16384 halves: mlp W, FRAGMENT order
#define WS_H16    317952       // 49152 halves: graph filter, FRAGMENT order
#define WS_FEAT1  416256       // 10240*128 halves: CNN features
#define WS_FEAT2  3037696      // 10240*128 halves: compressed features

struct PrepPtrs {
    const float* cw[5];
    const float* cb[5];
    const float* bg[5];
    const float* bb[5];
    const float* bm[5];
    const float* bv[5];
    const float* mlp_w;
    const float* gfh;
};

// ------------------------------------------------------------------
// Kernel 0: weight conversion + BN folding + fragment shuffling
// frag[((mt*NK + kc)*32 + lane)*16 + j] = W[m(lane)][k(j,lane)]
// ------------------------------------------------------------------
__global__ void prep_kernel(PrepPtrs p, _Float16* __restrict__ w16,
                            float* __restrict__ scale, float* __restrict__ shift,
                            _Float16* __restrict__ mw16, _Float16* __restrict__ h16) {
    const int tid = blockIdx.x * blockDim.x + threadIdx.x;
    const int nth = gridDim.x * blockDim.x;
    const int CINS[5]   = {8, 32, 32, 64, 64};   // layer 0 channel-padded
    const int COUTS[5]  = {32, 32, 64, 64, 128};
    const int KPADS[5]  = {96, 288, 288, 576, 576};
    const int WOFFS[5]  = {0, 3072, 12288, 30720, 67584};

    // conv weights -> f16, fragment order, HWC K-ordering, zero-padded
    for (int l = 0; l < 5; ++l) {
        const int NK  = KPADS[l] / 32;
        const int CIN = CINS[l];
        const int tot = COUTS[l] * KPADS[l];
        for (int i = tid; i < tot; i += nth) {
            const int j = i & 15, lane = (i >> 4) & 31;
            const int kc = (i >> 9) % NK, mt = (i >> 9) / NK;
            const int cout = mt * 16 + (lane & 15);
            const int k    = kc * 32 + ak_idx(j, lane);
            float v = 0.f;
            if (l == 0) {
                // k = rr*32 + cc*8 + cin ; valid iff rr<3, cc<3, cin<3
                const int rr = k / 32, t = k % 32, cc = t / 8, cin = t % 8;
                if (rr < 3 && cc < 3 && cin < 3)
                    v = p.cw[0][cout * 27 + cin * 9 + rr * 3 + cc];
            } else {
                // k = (rr*3+cc)*CIN + cin
                const int rc = k / CIN, cin = k % CIN;
                v = p.cw[l][cout * (CIN * 9) + cin * 9 + rc];
            }
            w16[WOFFS[l] + i] = (_Float16)v;
        }
    }
    // fused BN scale/shift (conv bias folded in)
    for (int i = tid; i < 320; i += nth) {
        int l, c;
        if (i < 32)       { l = 0; c = i; }
        else if (i < 64)  { l = 1; c = i - 32; }
        else if (i < 128) { l = 2; c = i - 64; }
        else if (i < 192) { l = 3; c = i - 128; }
        else              { l = 4; c = i - 192; }
        const float s = p.bg[l][c] * rsqrtf(p.bv[l][c] + 1e-5f);
        scale[i] = s;
        shift[i] = (p.cb[l][c] - p.bm[l][c]) * s + p.bb[l][c];
    }
    // mlp W -> fragment order (MT=8, NK=4)
    for (int i = tid; i < 128 * 128; i += nth) {
        const int j = i & 15, lane = (i >> 4) & 31;
        const int kc = (i >> 9) % 4, mt = (i >> 9) / 4;
        mw16[i] = (_Float16)p.mlp_w[(mt * 16 + (lane & 15)) * 128 + kc * 32 + ak_idx(j, lane)];
    }
    // gf_h [128,1,3,128] flat == [f][k], K = tap*128+g -> fragment order (MT=8, NK=12)
    for (int i = tid; i < 128 * 384; i += nth) {
        const int j = i & 15, lane = (i >> 4) & 31;
        const int kc = (i >> 9) % 12, mt = (i >> 9) / 12;
        h16[i] = (_Float16)p.gfh[(mt * 16 + (lane & 15)) * 384 + kc * 32 + ak_idx(j, lane)];
    }
}

// ------------------------------------------------------------------
// LDS helpers
// ------------------------------------------------------------------
static __device__ __forceinline__ void zero_lds(_Float16* p, int n16) {
    const v16h z = {};
    for (int i = threadIdx.x; i < n16; i += blockDim.x)
        *(v16h*)(p + i * 16) = z;
}

// ------------------------------------------------------------------
// Implicit-GEMM conv via WMMA on HWC zero-padded LDS activations.
// Work unit = 16 cols x 32 couts (two accumulators per B fragment).
// CIN>=32: B fragment = ONE contiguous 32B LDS vector load.
// CIN==8 : B fragment = one window row = two 16B-aligned v8h loads.
// ------------------------------------------------------------------
template <int CIN, int HH, int COUT, int KPAD, int WOFF, int SCOFF, int DPAD>
static __device__ __forceinline__ void conv_wmma(const _Float16* __restrict__ w16,
                                                 const float* __restrict__ scale,
                                                 const float* __restrict__ shift,
                                                 const _Float16* src, _Float16* dst) {
    constexpr int SW   = HH + 2;              // padded src width
    constexpr int NPOS = HH * HH;
    constexpr int NTOT = NPOS * IMG;
    constexpr int NT   = (NTOT + 15) / 16;
    constexpr int MT   = COUT / 16;
    constexpr int MGR  = MT / 2;
    constexpr int NK   = KPAD / 32;
    constexpr int DW   = HH + 2 * DPAD;       // dst width (padded iff DPAD)
    constexpr int CHPC = (CIN >= 32) ? CIN / 32 : 1;  // K-chunks per (rr,cc)
    const int lane = threadIdx.x & 31;
    const int wave = threadIdx.x >> 5;
    const int kb0  = (lane >> 4) << 4;

    for (int t = wave; t < NT * MGR; t += NWAVES) {
        const int nt = t / MGR, mg = t % MGR;
        const int n    = nt * 16 + (lane & 15);
        const bool nok = (n < NTOT);
        const int nn   = nok ? n : 0;
        const int img  = nn / NPOS, pos = nn % NPOS;
        const int oh   = pos / HH,  ow  = pos % HH;
        const _Float16* sb = src + (img * SW * SW + oh * SW + ow) * CIN;

        v8f acc0 = {}, acc1 = {};
#pragma unroll
        for (int kc = 0; kc < NK; ++kc) {
            v16h b;
            if (CIN >= 32) {
                const int rc = kc / CHPC;                 // (rr,cc) pair
                const int cinBase = (kc % CHPC) * 32;
                const int rr = rc / 3, cc = rc % 3;
                b = *(const v16h*)(sb + (rr * SW + cc) * CIN + cinBase + kb0);
            } else {
                // layer 0: K-chunk kc == window row rr; 16B-aligned row segment
                const _Float16* q = sb + kc * SW * CIN + kb0;
                const v8h lo = *(const v8h*)(q);
                const v8h hi = *(const v8h*)(q + 8);
#pragma unroll
                for (int e = 0; e < 8; ++e) { b[e] = lo[e]; b[e + 8] = hi[e]; }
            }
            const v16h a0 = *(const v16h*)(w16 + WOFF +
                                           (((mg * 2 + 0) * NK + kc) << 9) + (lane << 4));
            acc0 = __builtin_amdgcn_wmma_f32_16x16x32_f16(false, a0, false, b,
                                                          (short)0, acc0, false, false);
            const v16h a1 = *(const v16h*)(w16 + WOFF +
                                           (((mg * 2 + 1) * NK + kc) << 9) + (lane << 4));
            acc1 = __builtin_amdgcn_wmma_f32_16x16x32_f16(false, a1, false, b,
                                                          (short)0, acc1, false, false);
        }
        // vectorized epilogue: BN + ReLU + packed 16B stores (HWC: couts contiguous)
        const int rowb = (lane >> 4) << 3;
        const int c0 = (mg * 2 + 0) * 16 + rowb, c1 = (mg * 2 + 1) * 16 + rowb;
        const v8f sc0 = *(const v8f*)(scale + SCOFF + c0);
        const v8f sh0 = *(const v8f*)(shift + SCOFF + c0);
        const v8f sc1 = *(const v8f*)(scale + SCOFF + c1);
        const v8f sh1 = *(const v8f*)(shift + SCOFF + c1);
        v8h o0, o1;
#pragma unroll
        for (int r = 0; r < 8; ++r) {
            const float v0 = acc0[r] * sc0[r] + sh0[r];
            o0[r] = (_Float16)(v0 > 0.f ? v0 : 0.f);
            const float v1 = acc1[r] * sc1[r] + sh1[r];
            o1[r] = (_Float16)(v1 > 0.f ? v1 : 0.f);
        }
        if (nok) {
            _Float16* db = dst + ((img * DW + oh + DPAD) * DW + ow + DPAD) * COUT;
            *(v8h*)(db + c0) = o0;
            *(v8h*)(db + c1) = o1;
        }
    }
}

// 2x2 maxpool on HWC, 8 channels per vector op; dst optionally halo-padded
template <int C, int HH, int DPAD>
static __device__ __forceinline__ void maxpool2(const _Float16* src, _Float16* dst) {
    constexpr int OH  = HH / 2;
    constexpr int DW  = OH + 2 * DPAD;
    constexpr int C8  = C / 8;
    constexpr int TOT = IMG * OH * OH * C8;
    for (int i = threadIdx.x; i < TOT; i += blockDim.x) {
        const int cb  = (i % C8) * 8;
        const int x   = (i / C8) % OH;
        const int y   = (i / (C8 * OH)) % OH;
        const int img = i / (C8 * OH * OH);
        const _Float16* s = src + (img * HH * HH + 2 * y * HH + 2 * x) * C + cb;
        const v8h s00 = *(const v8h*)(s);
        const v8h s01 = *(const v8h*)(s + C);
        const v8h s10 = *(const v8h*)(s + HH * C);
        const v8h s11 = *(const v8h*)(s + HH * C + C);
        v8h o;
#pragma unroll
        for (int e = 0; e < 8; ++e) {
            const float m = fmaxf(fmaxf((float)s00[e], (float)s01[e]),
                                  fmaxf((float)s10[e], (float)s11[e]));
            o[e] = (_Float16)m;
        }
        *(v8h*)(dst + (img * DW * DW + (y + DPAD) * DW + (x + DPAD)) * C + cb) = o;
    }
}

__global__ void __launch_bounds__(128) cnn_kernel(const float* __restrict__ in,
                                                  const _Float16* __restrict__ w16,
                                                  const float* __restrict__ scale,
                                                  const float* __restrict__ shift,
                                                  _Float16* __restrict__ feat1) {
    __shared__ __attribute__((aligned(32))) _Float16 bufA[IMG * 11 * 11 * 32]; // 15488
    __shared__ __attribute__((aligned(32))) _Float16 bufB[IMG * 5 * 5 * 64];   // 6400

    const int ibase = blockIdx.x * IMG;
    __builtin_prefetch(in + ibase * 363, 0, 1);   // global_prefetch_b8

    // input CHW -> padded HWC8 [img][13][13][8] (ch 3..7 zero)
    zero_lds(bufB, 350);                          // 5600 halves: covers 5408 + row overrun
    __syncthreads();
    for (int i = threadIdx.x; i < IMG * 363; i += blockDim.x) {
        const int img = i / 363, r = i % 363;
        const int c = r / 121, pp = r % 121, y = pp / 11, x = pp % 11;
        bufB[(img * 169 + (y + 1) * 13 + (x + 1)) * 8 + c] = (_Float16)in[ibase * 363 + i];
    }
    __syncthreads();
    conv_wmma<8, 11, 32, 96, 0, 0, 0>(w16, scale, shift, bufB, bufA);   // [11][11][32]
    __syncthreads();
    zero_lds(bufB, 392);                          // 6272: padded [7][7][32]
    __syncthreads();
    maxpool2<32, 11, 1>(bufA, bufB);
    __syncthreads();
    zero_lds(bufA, 392);
    __syncthreads();
    conv_wmma<32, 5, 32, 288, 3072, 32, 1>(w16, scale, shift, bufB, bufA);   // padded out
    __syncthreads();
    conv_wmma<32, 5, 64, 288, 12288, 64, 0>(w16, scale, shift, bufA, bufB);  // [5][5][64]
    __syncthreads();
    zero_lds(bufA, 256);                          // 4096: padded [4][4][64]
    __syncthreads();
    maxpool2<64, 5, 1>(bufB, bufA);
    __syncthreads();
    zero_lds(bufB, 256);
    __syncthreads();
    conv_wmma<64, 2, 64, 576, 30720, 128, 1>(w16, scale, shift, bufA, bufB);  // padded out
    __syncthreads();
    conv_wmma<64, 2, 128, 576, 67584, 192, 0>(w16, scale, shift, bufB, bufA); // [2][2][128]
    __syncthreads();
    maxpool2<128, 2, 0>(bufA, bufB);              // -> bufB[img*128 + c]
    __syncthreads();
    for (int i = threadIdx.x; i < IMG * FEAT; i += blockDim.x)
        feat1[ibase * FEAT + i] = bufB[i];
}

// ------------------------------------------------------------------
// compressMLP: y[r,f] = relu(sum_g x[r,g] W[f,g] + b[f]); 10240x128x128
// ------------------------------------------------------------------
__global__ void __launch_bounds__(256) mlp_kernel(const _Float16* __restrict__ feat1,
                                                  const _Float16* __restrict__ mw,
                                                  const float* __restrict__ mb,
                                                  _Float16* __restrict__ feat2) {
    const int lane = threadIdx.x & 31;
    const int wave = threadIdx.x >> 5;
    const int rowBase = (blockIdx.x * 8 + wave) * 16;   // 640 ntiles of 16 rows
    const int r = rowBase + (lane & 15);
    const int kb0 = (lane >> 4) << 4;

    v16h bf[4];
#pragma unroll
    for (int kc = 0; kc < 4; ++kc)
        bf[kc] = *(const v16h*)(feat1 + r * FEAT + kc * 32 + kb0);

    for (int mt = 0; mt < 8; ++mt) {
        v8f acc = {};
#pragma unroll
        for (int kc = 0; kc < 4; ++kc) {
            const v16h a = *(const v16h*)(mw + ((mt * 4 + kc) << 9) + (lane << 4));
            acc = __builtin_amdgcn_wmma_f32_16x16x32_f16(false, a, false, bf[kc],
                                                         (short)0, acc, false, false);
        }
        const int fb  = mt * 16 + ((lane >> 4) << 3);
        const int col = rowBase + (lane & 15);
        const v8f bias = *(const v8f*)(mb + fb);
        v8h o;
#pragma unroll
        for (int rr = 0; rr < 8; ++rr) {
            const float v = acc[rr] + bias[rr];
            o[rr] = (_Float16)(v > 0.f ? v : 0.f);
        }
        *(v8h*)(feat2 + col * FEAT + fb) = o;   // contiguous in f: one 16B store
    }
}

// ------------------------------------------------------------------
// Graph filter + actions head: one wave per sample b.
// zst [n][K] and ylds [n][f] so all WMMA-side accesses are vector ops.
// ------------------------------------------------------------------
__global__ void __launch_bounds__(32) graph_kernel(const _Float16* __restrict__ feat2,
                                                   const float* __restrict__ S,
                                                   const _Float16* __restrict__ h16,
                                                   const float* __restrict__ gfb,
                                                   const float* __restrict__ aw,
                                                   const float* __restrict__ ab,
                                                   float* __restrict__ out) {
    __shared__ float zb0[FEAT * 10];
    __shared__ float zb1[FEAT * 10];
    __shared__ float Sl[100];
    __shared__ __attribute__((aligned(32))) float    ylds[16 * FEAT];  // [n][f]
    __shared__ __attribute__((aligned(32))) _Float16 zst[16 * 384];    // [n][tap*128+g]

    const int b = blockIdx.x;
    const int lane = threadIdx.x;

    for (int i = lane; i < 100; i += 32) Sl[i] = S[b * 100 + i];
    for (int i = lane; i < 16 * 384; i += 32) zst[i] = (_Float16)0.f;
    __syncthreads();

    // tap 0
    for (int i = lane; i < FEAT * 10; i += 32) {
        const int g = i / 10, n = i % 10;
        const float v = (float)feat2[(b * 10 + n) * FEAT + g];
        zb0[i] = v;
        zst[n * 384 + g] = (_Float16)v;
    }
    __syncthreads();
    // tap 1
    for (int i = lane; i < FEAT * 10; i += 32) {
        const int g = i / 10, n = i % 10;
        float s = 0.f;
#pragma unroll
        for (int m = 0; m < 10; ++m) s += zb0[g * 10 + m] * Sl[m * 10 + n];
        zb1[i] = s;
        zst[n * 384 + 128 + g] = (_Float16)s;
    }
    __syncthreads();
    // tap 2
    for (int i = lane; i < FEAT * 10; i += 32) {
        const int g = i / 10, n = i % 10;
        float s = 0.f;
#pragma unroll
        for (int m = 0; m < 10; ++m) s += zb1[g * 10 + m] * Sl[m * 10 + n];
        zst[n * 384 + 256 + g] = (_Float16)s;
    }
    __syncthreads();

    // Y = H(128x384) @ Zst(384x16) via WMMA, + bias, ReLU
    const int kb0 = (lane >> 4) << 4;
    for (int mt = 0; mt < 8; ++mt) {
        v8f acc = {};
#pragma unroll
        for (int kc = 0; kc < 12; ++kc) {
            const v16h a  = *(const v16h*)(h16 + ((mt * 12 + kc) << 9) + (lane << 4));
            const v16h bb = *(const v16h*)(zst + (lane & 15) * 384 + kc * 32 + kb0);
            acc = __builtin_amdgcn_wmma_f32_16x16x32_f16(false, a, false, bb,
                                                         (short)0, acc, false, false);
        }
        const int fb = mt * 16 + ((lane >> 4) << 3);
        const v8f bias = *(const v8f*)(gfb + fb);
        v8f o;
#pragma unroll
        for (int r = 0; r < 8; ++r) {
            const float v = acc[r] + bias[r];
            o[r] = v > 0.f ? v : 0.f;
        }
        *(v8f*)(ylds + (lane & 15) * FEAT + fb) = o;   // contiguous 32B store
    }
    __syncthreads();

    // logits[n,a] = y[n,:] . aw[a,:] + ab[a]
    for (int i = lane; i < 50; i += 32) {
        const int n = i / 5, a = i % 5;
        float s = ab[a];
        for (int f = 0; f < FEAT; ++f) s += ylds[n * FEAT + f] * aw[a * FEAT + f];
        out[(b * 10 + n) * 5 + a] = s;
    }
}

// ------------------------------------------------------------------
extern "C" void kernel_launch(void* const* d_in, const int* in_sizes, int n_in,
                              void* d_out, int out_size, void* d_ws, size_t ws_size,
                              hipStream_t stream) {
    (void)in_sizes; (void)n_in; (void)out_size; (void)ws_size;
    const float* inT = (const float*)d_in[0];
    const float* S   = (const float*)d_in[1];

    PrepPtrs pp;
    for (int l = 0; l < 5; ++l) {
        pp.cw[l] = (const float*)d_in[2 + l];
        pp.cb[l] = (const float*)d_in[7 + l];
        pp.bg[l] = (const float*)d_in[12 + l];
        pp.bb[l] = (const float*)d_in[17 + l];
        pp.bm[l] = (const float*)d_in[22 + l];
        pp.bv[l] = (const float*)d_in[27 + l];
    }
    pp.mlp_w = (const float*)d_in[32];
    pp.gfh   = (const float*)d_in[34];

    char* ws = (char*)d_ws;
    _Float16* W16   = (_Float16*)(ws + WS_W16);
    float*    SCALE = (float*)(ws + WS_SCALE);
    float*    SHIFT = (float*)(ws + WS_SHIFT);
    _Float16* MW16  = (_Float16*)(ws + WS_MW16);
    _Float16* H16   = (_Float16*)(ws + WS_H16);
    _Float16* FEAT1 = (_Float16*)(ws + WS_FEAT1);
    _Float16* FEAT2 = (_Float16*)(ws + WS_FEAT2);

    prep_kernel<<<512, 256, 0, stream>>>(pp, W16, SCALE, SHIFT, MW16, H16);
    cnn_kernel<<<BN_TOT / IMG, 128, 0, stream>>>(inT, W16, SCALE, SHIFT, FEAT1);
    mlp_kernel<<<BN_TOT / 16 / 8, 256, 0, stream>>>(FEAT1, MW16,
                                                    (const float*)d_in[33], FEAT2);
    graph_kernel<<<1024, 32, 0, stream>>>(FEAT2, S, H16,
                                          (const float*)d_in[35],
                                          (const float*)d_in[36],
                                          (const float*)d_in[37],
                                          (float*)d_out);
}
